// WindowAttention_9732395892958
// MI455X (gfx1250) — matrix operand order repaired and checked by reading, not compiled, not run on hardware
//
#include <hip/hip_runtime.h>
#include <cstdint>

typedef __attribute__((ext_vector_type(16))) _Float16 v16h;
typedef __attribute__((ext_vector_type(8)))  _Float16 v8h;
typedef __attribute__((ext_vector_type(4)))  _Float16 v4h;
typedef __attribute__((ext_vector_type(8)))  float    v8f;

#define NH   12
#define DH   32
#define WSQ  49
#define DIMS 384
#define NWIN 64
#define QK_SCALE 0.17677669529663687f

// LDS row strides (elements); all multiples of 8 halves -> 16B-aligned vectors
#define XS_S 392
#define WT_S 392
#define QK_S 40
#define VT_S 72
#define SM_S 68
#define AT_S 72
#define OS_S 392
#define MK_S 52
#define PANEL (96 * DIMS)   // one staged weight panel, in halves

// ---------------- WMMA fragment helpers (CDNA5 16x16x32 f16) ----------------
// A (16x32, MxK): lane<16 -> M=lane, K = {0..7, 16..23}; lane>=16 -> K = {8..15, 24..31}
__device__ __forceinline__ v16h load_a_frag(const _Float16* base, int stride,
                                            int row0, int col0, int lane) {
    const int l = lane & 15, hi = lane >> 4;
    const _Float16* p = base + (row0 + l) * stride + col0 + hi * 8;
    v8h x0 = *(const v8h*)(p);
    v8h x1 = *(const v8h*)(p + 16);
    v16h a;
#pragma unroll
    for (int i = 0; i < 8; ++i) { a[i] = x0[i]; a[i + 8] = x1[i]; }
    return a;
}

// B (32x16, KxN): n = lane&15, K = {0..7, 16..23} (lane<16) / {8..15, 24..31} (lane>=16), same as A.
// `base` stores B n-major: base[n*stride + k].
__device__ __forceinline__ v16h load_b_frag(const _Float16* base, int stride,
                                            int nrow0, int k0, int lane) {
    const int l = lane & 15, hi = lane >> 4;
    const _Float16* p = base + (nrow0 + l) * stride + k0 + hi * 8;
    v8h x0 = *(const v8h*)(p);
    v8h x1 = *(const v8h*)(p + 16);
    v16h b;
#pragma unroll
    for (int i = 0; i < 8; ++i) { b[i] = x0[i]; b[i + 8] = x1[i]; }
    return b;
}

__device__ __forceinline__ v8f wmma_f16(v16h a, v16h b, v8f c) {
    return __builtin_amdgcn_wmma_f32_16x16x32_f16(false, a, false, b,
                                                  (short)0, c, false, false);
}

__device__ __forceinline__ v8f zero8f() {
    v8f z;
#pragma unroll
    for (int i = 0; i < 8; ++i) z[i] = 0.0f;
    return z;
}

// ------- async global->LDS staging of a 96x384 f16 weight panel (issue only) -------
// 4608 b128 transfers; 256 threads -> exactly 18 per thread (all lanes active).
__device__ __forceinline__ void stage_issue_96x384(_Float16* dst,
                                                   const _Float16* __restrict__ src,
                                                   int tid) {
    for (int idx = tid; idx < 96 * 48; idx += 256) {
        const int row = idx / 48, seg = idx % 48;
        unsigned lds = (unsigned)(uintptr_t)(dst + row * WT_S + seg * 8);
        const _Float16* g = src + row * 384 + seg * 8;
        asm volatile("global_load_async_to_lds_b128 %0, %1, off"
                     :: "v"(lds), "v"(g) : "memory");
    }
}
// Wait until only the most recent panel (18 per-thread transfers) is still in flight.
__device__ __forceinline__ void async_wait_prev() {
    asm volatile("s_wait_asynccnt 18" ::: "memory");
}
__device__ __forceinline__ void async_wait_all() {
    asm volatile("s_wait_asynccnt 0" ::: "memory");
}

// ------------- prep: transpose + f16-convert weights into workspace -------------
// WTperm[h][r][k], r in 0..95 = (q cols 0..31 | k cols | v cols) of head h
// WoT[c][k] = w_out[k][c]
__global__ void swin_prep_kernel(const float* __restrict__ w_qkv,
                                 const float* __restrict__ w_out,
                                 _Float16* __restrict__ WTperm,
                                 _Float16* __restrict__ WoT) {
    const int tid = blockIdx.x * blockDim.x + threadIdx.x;
    const int total1 = NH * PANEL;
    if (tid < total1) {
        const int h = tid / PANEL;
        const int rem = tid % PANEL;
        const int r = rem / DIMS;
        const int kk = rem % DIMS;
        const int col = (r < 32) ? (h * 32 + r)
                      : (r < 64) ? (DIMS + h * 32 + (r - 32))
                                 : (2 * DIMS + h * 32 + (r - 64));
        WTperm[tid] = (_Float16)w_qkv[kk * (3 * DIMS) + col];
    } else if (tid < total1 + DIMS * DIMS) {
        const int t = tid - total1;
        const int c = t / DIMS, kk = t % DIMS;
        WoT[t] = (_Float16)w_out[kk * DIMS + c];
    }
}

// ---------------------------- fused window attention ----------------------------
__global__ __launch_bounds__(256, 1)
void swin_attn_kernel(const float* __restrict__ x,
                      const float* __restrict__ mask,
                      const float* __restrict__ bias_table,
                      const float* __restrict__ b_out,
                      const _Float16* __restrict__ WTperm,
                      const _Float16* __restrict__ WoT,
                      float* __restrict__ out) {
    __shared__ __align__(16) _Float16 xs[64 * XS_S];       // x, f16, rows padded to 64
    __shared__ __align__(16) _Float16 wt[2][96 * WT_S];    // double-buffered weight panel
    __shared__ __align__(16) _Float16 qs[64 * QK_S];       // q (scaled), row-major
    __shared__ __align__(16) _Float16 ks[64 * QK_S];       // k, row-major (B for QK^T)
    __shared__ __align__(16) _Float16 vt[32 * VT_S];       // v^T: [d][j] (B for AV)
    __shared__ __align__(16) float    sims[64 * SM_S];     // logits f32
    __shared__ __align__(16) _Float16 at[64 * AT_S];       // attn probs f16
    __shared__ __align__(16) _Float16 os[64 * OS_S];       // concat head outputs
    __shared__ __align__(16) float    maskl[WSQ * MK_S];
    __shared__ __align__(16) float    biasl[169 * NH];

    const int tid  = threadIdx.x;
    const int lane = tid & 31;
    const int wave = tid >> 5;
    const int b    = blockIdx.x;
    const int l    = lane & 15;
    const int hi   = lane >> 4;

    // Kick off the DMA of head 0's weight panel before anything else.
    stage_issue_96x384(wt[0], WTperm, tid);

    // ---- phase 0: stage x (f32->f16), mask, bias; zero paddings ----
    {
        const float4* xsrc = (const float4*)(x + (size_t)b * WSQ * DIMS);
        for (int i4 = tid; i4 < WSQ * DIMS / 4; i4 += 256) {
            const int e = i4 * 4, r = e / DIMS, c = e % DIMS;
            const float4 v = xsrc[i4];
            v4h hh;
            hh[0] = (_Float16)v.x; hh[1] = (_Float16)v.y;
            hh[2] = (_Float16)v.z; hh[3] = (_Float16)v.w;
            *(v4h*)(xs + r * XS_S + c) = hh;
        }
        for (int i = tid; i < 15 * DIMS; i += 256)
            xs[(WSQ + i / DIMS) * XS_S + (i % DIMS)] = (_Float16)0.0f;
        const float* msrc = mask + (size_t)(b % NWIN) * WSQ * WSQ;
        for (int i = tid; i < WSQ * WSQ; i += 256)
            maskl[(i / WSQ) * MK_S + (i % WSQ)] = msrc[i];
        for (int i = tid; i < 169 * NH; i += 256) biasl[i] = bias_table[i];
        for (int i = tid; i < 64 * AT_S; i += 256) at[i] = (_Float16)0.0f;
    }
    __syncthreads();

    for (int h = 0; h < NH; ++h) {
        const _Float16* wtc = wt[h & 1];
        // Prefetch the next panel (next head, or first projection chunk).
        if (h < NH - 1) stage_issue_96x384(wt[(h + 1) & 1], WTperm + (size_t)(h + 1) * PANEL, tid);
        else            stage_issue_96x384(wt[0], WoT, tid);
        async_wait_prev();           // current panel done; next still in flight
        __syncthreads();

        // ---- phase 2: q,k,v = x @ W. Each wave owns one (mt,nt) tile of all
        // three matrices; A fragments are loaded once and feed 3 WMMAs.
        {
            const int mt = wave >> 1, nt = wave & 1;
            v8f aq = zero8f(), ak = zero8f(), av = zero8f();
#pragma unroll
            for (int kb = 0; kb < 12; ++kb) {
                v16h a  = load_a_frag(xs, XS_S, mt * 16, kb * 32, lane);
                v16h bq = load_b_frag(wtc, WT_S,      nt * 16, kb * 32, lane);
                v16h bk = load_b_frag(wtc, WT_S, 32 + nt * 16, kb * 32, lane);
                v16h bv = load_b_frag(wtc, WT_S, 64 + nt * 16, kb * 32, lane);
                aq = wmma_f16(a, bq, aq);
                ak = wmma_f16(a, bk, ak);
                av = wmma_f16(a, bv, av);
            }
            const int n = nt * 16 + l;
#pragma unroll
            for (int r = 0; r < 8; ++r) {
                const int m = mt * 16 + hi * 8 + r;
                qs[m * QK_S + n] = (_Float16)(aq[r] * QK_SCALE);
                ks[m * QK_S + n] = (_Float16)ak[r];
                vt[n * VT_S + m] = (_Float16)av[r];   // store v transposed
            }
        }
        __syncthreads();

        // ---- phase 3: sim = q @ k^T (16 tiles; one K-step since DH=32).
        // Each wave: one A fragment, two N-tiles.
        {
            const int mt  = wave >> 1;
            const int nt0 = (wave & 1) * 2;
            v16h a = load_a_frag(qs, QK_S, mt * 16, 0, lane);
#pragma unroll
            for (int jj = 0; jj < 2; ++jj) {
                v16h bb = load_b_frag(ks, QK_S, (nt0 + jj) * 16, 0, lane);
                v8f acc = zero8f();
                acc = wmma_f16(a, bb, acc);
#pragma unroll
                for (int r = 0; r < 8; ++r)
                    sims[(mt * 16 + hi * 8 + r) * SM_S + (nt0 + jj) * 16 + l] = acc[r];
            }
        }
        __syncthreads();

        // ---- phase 4: softmax rows (+rel-pos bias +window mask), 4 lanes/row ----
        {
            const int i   = tid >> 2;     // row
            const int sub = tid & 3;
            if (i < WSQ) {
                const int ci = (i / 7) * 13 + (i % 7);
                float mx = -3.0e38f;
                for (int j = sub; j < WSQ; j += 4) {
                    const int p = 48 - j;
                    const int idx = ci + (p / 7) * 13 + (p % 7);
                    float v = sims[i * SM_S + j] + biasl[idx * NH + h]
                            + maskl[i * MK_S + j];
                    sims[i * SM_S + j] = v;
                    mx = fmaxf(mx, v);
                }
                mx = fmaxf(mx, __shfl_xor(mx, 1, 32));
                mx = fmaxf(mx, __shfl_xor(mx, 2, 32));
                float sum = 0.0f;
                for (int j = sub; j < WSQ; j += 4) {
                    const float e = __expf(sims[i * SM_S + j] - mx);
                    sum += e;
                    sims[i * SM_S + j] = e;
                }
                sum += __shfl_xor(sum, 1, 32);
                sum += __shfl_xor(sum, 2, 32);
                const float rs = 1.0f / sum;
                for (int j = sub; j < WSQ; j += 4)
                    at[i * AT_S + j] = (_Float16)(sims[i * SM_S + j] * rs);
            }
        }
        __syncthreads();

        // ---- phase 5: out_h = attn @ v (8 tiles, 1 per wave) ----
        {
            const int mt = wave >> 1, nt = wave & 1;
            v8f acc = zero8f();
#pragma unroll
            for (int kb = 0; kb < 2; ++kb) {
                v16h a  = load_a_frag(at, AT_S, mt * 16, kb * 32, lane);
                v16h bb = load_b_frag(vt, VT_S, nt * 16, kb * 32, lane);
                acc = wmma_f16(a, bb, acc);
            }
#pragma unroll
            for (int r = 0; r < 8; ++r) {
                const int m = mt * 16 + hi * 8 + r;
                os[m * OS_S + h * DH + nt * 16 + l] = (_Float16)acc[r];
            }
        }
        __syncthreads();
    }

    // ---- phase 6: out = concat @ w_out + b_out (4 chunks of 96 cols) ----
    // Each wave: mt = wave>>1, three N-tiles ntb..ntb+2; A loaded once per kb.
    for (int cb = 0; cb < 4; ++cb) {
        const _Float16* wtc = wt[cb & 1];
        if (cb < 3) {
            stage_issue_96x384(wt[(cb + 1) & 1], WoT + (size_t)(cb + 1) * PANEL, tid);
            async_wait_prev();
        } else {
            async_wait_all();
        }
        __syncthreads();

        {
            const int mt  = wave >> 1;
            const int ntb = (wave & 1) * 3;
            v8f acc[3];
#pragma unroll
            for (int t = 0; t < 3; ++t) acc[t] = zero8f();
#pragma unroll
            for (int kb = 0; kb < 12; ++kb) {
                v16h a = load_a_frag(os, OS_S, mt * 16, kb * 32, lane);
#pragma unroll
                for (int t = 0; t < 3; ++t) {
                    v16h bb = load_b_frag(wtc, WT_S, (ntb + t) * 16, kb * 32, lane);
                    acc[t] = wmma_f16(a, bb, acc[t]);
                }
            }
#pragma unroll
            for (int t = 0; t < 3; ++t) {
                const int col = cb * 96 + (ntb + t) * 16 + l;
                const float bo = b_out[col];
#pragma unroll
                for (int r = 0; r < 8; ++r) {
                    const int m = mt * 16 + hi * 8 + r;
                    if (m < WSQ)
                        out[((size_t)b * WSQ + m) * DIMS + col] = acc[t][r] + bo;
                }
            }
        }
        __syncthreads();
    }
}

extern "C" void kernel_launch(void* const* d_in, const int* in_sizes, int n_in,
                              void* d_out, int out_size, void* d_ws, size_t ws_size,
                              hipStream_t stream) {
    const float* x          = (const float*)d_in[0]; // (2048, 49, 384)
    const float* mask       = (const float*)d_in[1]; // (64, 49, 49)
    const float* w_qkv      = (const float*)d_in[2]; // (384, 1152)
    const float* bias_table = (const float*)d_in[3]; // (169, 12)
    const float* w_out      = (const float*)d_in[4]; // (384, 384)
    const float* b_out      = (const float*)d_in[5]; // (384,)

    _Float16* WTperm = (_Float16*)d_ws;                                   // 12*96*384 f16
    _Float16* WoT    = (_Float16*)((char*)d_ws + (size_t)NH * PANEL * 2); // 384*384 f16

    const int prep_total = NH * PANEL + DIMS * DIMS;
    swin_prep_kernel<<<(prep_total + 255) / 256, 256, 0, stream>>>(
        w_qkv, w_out, WTperm, WoT);

    swin_attn_kernel<<<2048, 256, 0, stream>>>(
        x, mask, bias_table, b_out, WTperm, WoT, (float*)d_out);
}
